// GradientNormLayer_979252543506
// MI455X (gfx1250) — compile-verified, matrix operand order checked by reading, not run on hardware
//
#include <hip/hip_runtime.h>
#include <math.h>

// Reference shapes
#define B_DIM 1024
#define D_DIM 512
#define C_DIM 512

typedef __attribute__((ext_vector_type(16))) _Float16 v16h;
typedef __attribute__((ext_vector_type(8)))  _Float16 v8h;
typedef __attribute__((ext_vector_type(4)))  _Float16 v4h;
typedef __attribute__((ext_vector_type(2)))  _Float16 v2h;
typedef __attribute__((ext_vector_type(8)))  float    v8f;

// ---------------------------------------------------------------------------
// Prep kernel 1: X [1024][512] f32 -> f16 row-major (A fragments become two
// contiguous b128 loads per lane; no VALU converts in the GEMM loop).
// ---------------------------------------------------------------------------
__global__ __launch_bounds__(256) void pack_x_kernel(const float* __restrict__ X,
                                                     _Float16* __restrict__ Xh) {
    const int i = blockIdx.x * 256 + threadIdx.x;        // 0..131071 (x4 floats)
    const float4 v = reinterpret_cast<const float4*>(X)[i];
    v4h h;
    h[0] = (_Float16)v.x; h[1] = (_Float16)v.y;
    h[2] = (_Float16)v.z; h[3] = (_Float16)v.w;
    reinterpret_cast<v4h*>(Xh)[i] = h;
}

// ---------------------------------------------------------------------------
// Prep kernel 2: W [512][512] f32 -> packed f16 in exact WMMA B-fragment dword
// order: Wp[b][col][kp] = pack(W[32b+2kp][col], W[32b+2kp+1][col]),
// b = K-block (0..15), kp = K-pair within block (0..15).
// A lane's B fragment = dwords kp {4h..4h+3} and {8+4h..11+4h} -> two b128s.
// ---------------------------------------------------------------------------
__global__ __launch_bounds__(256) void pack_w_kernel(const float* __restrict__ W,
                                                     unsigned int* __restrict__ Wp) {
    const int e   = blockIdx.x * 256 + threadIdx.x;      // 0..131071
    const int col = e & 511;
    const int kp  = (e >> 9) & 15;
    const int b   = e >> 13;
    const int k   = b * 32 + kp * 2;
    const float w0 = W[k * C_DIM + col];
    const float w1 = W[(k + 1) * C_DIM + col];
    union { v2h h; unsigned int u; } p;
    p.h[0] = (_Float16)w0; p.h[1] = (_Float16)w1;
    Wp[b * 8192 + col * 16 + kp] = p.u;
}

// ---------------------------------------------------------------------------
// Main kernel: one WG = 16 rows x full C=512. 8 wave32 waves x 4 N-tiles.
// K-loop: 2 A b128 loads + 8 B b128 loads (all hoisted, independent regs),
// then 4 v_wmma_f32_16x16x32_f16 — loads overlap matrix ops.
// Then fused softmax + grad norm: out = sqrt((||x||^2+1) * ||p - y||^2).
// ---------------------------------------------------------------------------
__global__ __launch_bounds__(256) void
gradnorm_wmma_kernel(const _Float16* __restrict__ Xh,
                     const unsigned int* __restrict__ Wp,
                     const float* __restrict__ X,
                     const float* __restrict__ Y,
                     const float* __restrict__ bias,
                     float* __restrict__ out)
{
    __shared__ float ldsZ[16 * C_DIM];   // 32 KB logits tile

    const int tid     = threadIdx.x;
    const int wave    = tid >> 5;
    const int lane    = tid & 31;
    const int rowBase = blockIdx.x * 16;

    const int half16  = lane >> 4;       // 0: lanes 0-15, 1: lanes 16-31
    const int l15     = lane & 15;
    const int kSel    = half16 * 8;      // fragment K sub-offset
    const int aRow    = rowBase + l15;
    const int colBase = wave * 64;

    v8f acc[4] = {};
    const _Float16* aRowPtr = Xh + aRow * D_DIM;

    union Frag { v16h h; uint4 q[2]; };

    #pragma unroll 4
    for (int step = 0; step < 16; ++step) {
        const int k0 = step * 32;

        if (step < 15)   // WGP-scope prefetch of next packed-W block
            __builtin_prefetch(Wp + (step + 1) * 8192 + (colBase + l15) * 16, 0, 3);

        // A fragment: two contiguous b128 loads of pre-packed f16.
        Frag a;
        a.q[0] = *reinterpret_cast<const uint4*>(aRowPtr + k0 + kSel);
        a.q[1] = *reinterpret_cast<const uint4*>(aRowPtr + k0 + kSel + 16);

        const unsigned int* wblk = Wp + step * 8192;

        // Hoist all 4 B fragments into independent registers so the WMMAs
        // below retire with partial loadcnt waits instead of wait-0 each.
        Frag bfr[4];
        #pragma unroll
        for (int t = 0; t < 4; ++t) {
            const int col = colBase + 16 * t + l15;
            const uint4* bp = reinterpret_cast<const uint4*>(wblk + col * 16);
            bfr[t].q[0] = bp[half16];       // dwords kp = 4*half16 .. +3
            bfr[t].q[1] = bp[2 + half16];   // dwords kp = 8+4*half16 .. +3
        }

        #pragma unroll
        for (int t = 0; t < 4; ++t) {
            // (neg_a, A, neg_b, B, c_mod, C, reuse_a, reuse_b)
            acc[t] = __builtin_amdgcn_wmma_f32_16x16x32_f16(
                false, a.h, false, bfr[t].h, (short)0, acc[t], false, false);
        }
    }

    // Epilogue: bias + spill logits. D layout: reg j, lane L -> M=j+8*(L>=16), N=L&15.
    #pragma unroll
    for (int t = 0; t < 4; ++t) {
        const int col = colBase + 16 * t + l15;
        const float bv = bias[col];
        #pragma unroll
        for (int j = 0; j < 8; ++j)
            ldsZ[(j + 8 * half16) * C_DIM + col] = acc[t][j] + bv;
    }
    __syncthreads();

    // Fused softmax + gradient norm (16 threads per row, half-wave shuffles).
    const int row  = tid >> 4;
    const int sub  = tid & 15;
    const int gRow = rowBase + row;
    const float* zrow = &ldsZ[row * C_DIM];

    float m = -INFINITY;
    #pragma unroll 4
    for (int i = sub; i < C_DIM; i += 16) m = fmaxf(m, zrow[i]);
    #pragma unroll
    for (int off = 8; off >= 1; off >>= 1) m = fmaxf(m, __shfl_xor(m, off, 32));

    float s = 0.0f;
    #pragma unroll 4
    for (int i = sub; i < C_DIM; i += 16) s += __expf(zrow[i] - m);
    #pragma unroll
    for (int off = 8; off >= 1; off >>= 1) s += __shfl_xor(s, off, 32);
    const float inv_s = 1.0f / s;

    float g2 = 0.0f;
    #pragma unroll 4
    for (int i = sub; i < C_DIM; i += 16) {
        const float p = __expf(zrow[i] - m) * inv_s;
        const float d = p - Y[gRow * C_DIM + i];
        g2 += d * d;
    }
    #pragma unroll
    for (int off = 8; off >= 1; off >>= 1) g2 += __shfl_xor(g2, off, 32);

    float xn = 0.0f;
    #pragma unroll 4
    for (int i = sub; i < D_DIM; i += 16) {
        const float v = X[gRow * D_DIM + i];
        xn += v * v;
    }
    #pragma unroll
    for (int off = 8; off >= 1; off >>= 1) xn += __shfl_xor(xn, off, 32);

    if (sub == 0) out[gRow] = sqrtf((xn + 1.0f) * g2);
}

// ---------------------------------------------------------------------------
// Fallback (self-contained, LDS-staged) if ws is too small.
// ---------------------------------------------------------------------------
__global__ __launch_bounds__(256) void
gradnorm_fused_fallback(const float* __restrict__ X,
                        const float* __restrict__ Y,
                        const float* __restrict__ W,
                        const float* __restrict__ bias,
                        float* __restrict__ out)
{
    __shared__ __align__(16) unsigned char smem[64 * 1024];
    _Float16* wslab = reinterpret_cast<_Float16*>(smem);
    float*    ldsZ  = reinterpret_cast<float*>(smem);

    const int tid     = threadIdx.x;
    const int wave    = tid >> 5;
    const int lane    = tid & 31;
    const int rowBase = blockIdx.x * 16;
    const int half16  = lane >> 4;
    const int l15     = lane & 15;
    const int kSel    = half16 * 8;
    const int kpBase  = half16 * 4;
    const int aRow    = rowBase + l15;
    const int colBase = wave * 64;

    auto fillW = [&](int k0, int buf) {
        _Float16* dst = wslab + buf * (16 * 1024);
        #pragma unroll
        for (int it = 0; it < 8; ++it) {
            const int e  = tid + it * 256;
            const int kp = e >> 7;
            const int cc = (e & 127) << 2;
            const float4 r0 = *reinterpret_cast<const float4*>(&W[(k0 + 2 * kp    ) * C_DIM + cc]);
            const float4 r1 = *reinterpret_cast<const float4*>(&W[(k0 + 2 * kp + 1) * C_DIM + cc]);
            v8h pk;
            pk[0] = (_Float16)r0.x; pk[1] = (_Float16)r1.x;
            pk[2] = (_Float16)r0.y; pk[3] = (_Float16)r1.y;
            pk[4] = (_Float16)r0.z; pk[5] = (_Float16)r1.z;
            pk[6] = (_Float16)r0.w; pk[7] = (_Float16)r1.w;
            *reinterpret_cast<v8h*>(&dst[kp * 1024 + cc * 2]) = pk;
        }
    };

    v8f acc[4] = {};
    fillW(0, 0);
    for (int step = 0; step < 16; ++step) {
        __syncthreads();
        const int k0 = step * 32;
        if (step < 15) fillW(k0 + 32, (step + 1) & 1);

        const float4* ap0 = reinterpret_cast<const float4*>(&X[aRow * D_DIM + k0 + kSel]);
        const float4* ap1 = reinterpret_cast<const float4*>(&X[aRow * D_DIM + k0 + kSel + 16]);
        float4 a0 = ap0[0], a1 = ap0[1], a2 = ap1[0], a3 = ap1[1];
        v16h afrag;
        afrag[0]  = (_Float16)a0.x; afrag[1]  = (_Float16)a0.y;
        afrag[2]  = (_Float16)a0.z; afrag[3]  = (_Float16)a0.w;
        afrag[4]  = (_Float16)a1.x; afrag[5]  = (_Float16)a1.y;
        afrag[6]  = (_Float16)a1.z; afrag[7]  = (_Float16)a1.w;
        afrag[8]  = (_Float16)a2.x; afrag[9]  = (_Float16)a2.y;
        afrag[10] = (_Float16)a2.z; afrag[11] = (_Float16)a2.w;
        afrag[12] = (_Float16)a3.x; afrag[13] = (_Float16)a3.y;
        afrag[14] = (_Float16)a3.z; afrag[15] = (_Float16)a3.w;

        const _Float16* wb = wslab + (step & 1) * (16 * 1024);
        #pragma unroll
        for (int t = 0; t < 4; ++t) {
            const int col2 = (colBase + 16 * t + l15) * 2;
            v16h bfrag;
            #pragma unroll
            for (int j = 0; j < 4; ++j) {
                const v2h lo = *reinterpret_cast<const v2h*>(&wb[(kpBase + j    ) * 1024 + col2]);
                const v2h hi = *reinterpret_cast<const v2h*>(&wb[(kpBase + 8 + j) * 1024 + col2]);
                bfrag[2 * j] = lo[0]; bfrag[2 * j + 1] = lo[1];
                bfrag[8 + 2 * j] = hi[0]; bfrag[9 + 2 * j] = hi[1];
            }
            acc[t] = __builtin_amdgcn_wmma_f32_16x16x32_f16(
                false, afrag, false, bfrag, (short)0, acc[t], false, false);
        }
    }

    #pragma unroll
    for (int t = 0; t < 4; ++t) {
        const int col = colBase + 16 * t + l15;
        const float bv = bias[col];
        #pragma unroll
        for (int j = 0; j < 8; ++j)
            ldsZ[(j + 8 * half16) * C_DIM + col] = acc[t][j] + bv;
    }
    __syncthreads();

    const int row  = tid >> 4;
    const int sub  = tid & 15;
    const int gRow = rowBase + row;
    const float* zrow = &ldsZ[row * C_DIM];

    float m = -INFINITY;
    for (int i = sub; i < C_DIM; i += 16) m = fmaxf(m, zrow[i]);
    #pragma unroll
    for (int off = 8; off >= 1; off >>= 1) m = fmaxf(m, __shfl_xor(m, off, 32));
    float s = 0.0f;
    for (int i = sub; i < C_DIM; i += 16) s += __expf(zrow[i] - m);
    #pragma unroll
    for (int off = 8; off >= 1; off >>= 1) s += __shfl_xor(s, off, 32);
    const float inv_s = 1.0f / s;
    float g2 = 0.0f;
    for (int i = sub; i < C_DIM; i += 16) {
        const float p = __expf(zrow[i] - m) * inv_s;
        const float d = p - Y[gRow * C_DIM + i];
        g2 += d * d;
    }
    #pragma unroll
    for (int off = 8; off >= 1; off >>= 1) g2 += __shfl_xor(g2, off, 32);
    float xn = 0.0f;
    for (int i = sub; i < D_DIM; i += 16) { const float v = X[gRow * D_DIM + i]; xn += v * v; }
    #pragma unroll
    for (int off = 8; off >= 1; off >>= 1) xn += __shfl_xor(xn, off, 32);
    if (sub == 0) out[gRow] = sqrtf((xn + 1.0f) * g2);
}

extern "C" void kernel_launch(void* const* d_in, const int* in_sizes, int n_in,
                              void* d_out, int out_size, void* d_ws, size_t ws_size,
                              hipStream_t stream) {
    // setup_inputs order: x [B,D] f32, y_true [B,C] f32, W [D,C] f32, b [C] f32
    const float* x = (const float*)d_in[0];
    const float* y = (const float*)d_in[1];
    const float* W = (const float*)d_in[2];
    const float* b = (const float*)d_in[3];
    float* out = (float*)d_out;          // [B,1] f32

    (void)in_sizes; (void)n_in; (void)out_size;

    const size_t xh_bytes = (size_t)B_DIM * D_DIM * sizeof(_Float16);   // 1 MB
    const size_t wp_bytes = (size_t)D_DIM * C_DIM * sizeof(_Float16);   // 512 KB

    if (ws_size >= xh_bytes + wp_bytes) {
        _Float16*     Xh = (_Float16*)d_ws;
        unsigned int* Wp = (unsigned int*)((char*)d_ws + xh_bytes);
        pack_x_kernel<<<(B_DIM * D_DIM / 4) / 256, 256, 0, stream>>>(x, Xh);
        pack_w_kernel<<<(D_DIM * C_DIM / 2) / 256, 256, 0, stream>>>(W, Wp);
        gradnorm_wmma_kernel<<<B_DIM / 16, 256, 0, stream>>>(Xh, Wp, x, y, b, out);
    } else {
        gradnorm_fused_fallback<<<B_DIM / 16, 256, 0, stream>>>(x, y, W, b, out);
    }
}
